// SCIFARNetworkTorch_38534446580001
// MI455X (gfx1250) — matrix operand order repaired and checked by reading, not compiled, not run on hardware
//
#include <hip/hip_runtime.h>

// ---------------------------------------------------------------------------
// S4D network: encode -> 4x (S4D causal conv + FF/ReLU) -> decode(last pos)
// MI455X / gfx1250: wave32, WMMA f32_16x16x32_bf16, async LDS copy, TDM.
// ---------------------------------------------------------------------------

#define B_    32
#define L_    1024
#define DIN_  3
#define H_    512
#define NL_   4
#define NS_   64
#define DOUT_ 10
#define BL_   (B_ * L_)

typedef __attribute__((ext_vector_type(16))) __bf16       v16bf;
typedef __attribute__((ext_vector_type(8)))  float        v8f;
typedef __attribute__((ext_vector_type(4)))  unsigned int u32x4;

union Frag  { v16bf v; u32x4 q[2]; };
union BFrag { v16bf v; unsigned int u[8]; };

#if defined(__has_builtin)
#if __has_builtin(__builtin_amdgcn_global_load_async_to_lds_b128) && \
    __has_builtin(__builtin_amdgcn_s_wait_asynccnt)
#define HAVE_ASYNC_LDS 1
#endif
#if __has_builtin(__builtin_amdgcn_tensor_load_to_lds) && \
    __has_builtin(__builtin_amdgcn_s_wait_tensorcnt)
#define HAVE_TDM 1
#endif
#endif

#ifdef HAVE_ASYNC_LDS
// Exact pointee type from the compiler diagnostic: 16-byte int vector.
typedef int v4i __attribute__((vector_size(4 * sizeof(int))));
typedef __attribute__((address_space(1))) v4i gv4i;   // global (AS1, "__device__")
typedef __attribute__((address_space(3))) v4i sv4i;   // LDS (AS3)
#endif

#ifdef HAVE_TDM
typedef unsigned int u32x4_t __attribute__((vector_size(16)));
typedef int          i32x8_t __attribute__((vector_size(32)));
typedef int          i32x4_t __attribute__((vector_size(16)));
#endif

__device__ __forceinline__ v8f wmma_bf16(v16bf a, v16bf b, v8f c) {
  // D = A(16x32) * B(32x16) + C, f32 accumulate
  return __builtin_amdgcn_wmma_f32_16x16x32_bf16(
      /*neg_a=*/false, a, /*neg_b=*/false, b,
      /*c_mod=*/(short)0, c, /*reuse_a=*/false, /*reuse_b=*/false);
}

// --------------------------- encode: x @ W_enc ------------------------------
__global__ void encode_kernel(const float* __restrict__ x,
                              const float* __restrict__ W_enc,
                              __bf16* __restrict__ h0) {
  int idx = blockIdx.x * blockDim.x + threadIdx.x;   // over BL_*H_
  if (idx >= BL_ * H_) return;
  int hh = idx & (H_ - 1);
  int bl = idx >> 9;
  const float* xp = x + (size_t)bl * DIN_;
  float s = xp[0] * W_enc[0 * H_ + hh]
          + xp[1] * W_enc[1 * H_ + hh]
          + xp[2] * W_enc[2 * H_ + hh];
  h0[idx] = (__bf16)s;
}

// ---------------- per-layer discretization coefficients ---------------------
// dtA and CB are contiguous in the workspace so TDM can fetch both rows as one
// 2x64 f32 tile (row stride = H_*NS_ elements).
__global__ void coef_kernel(const float* __restrict__ log_dt,
                            const float* __restrict__ A_log,
                            const float* __restrict__ C,
                            float* __restrict__ dtA, float* __restrict__ CB,
                            int li) {
  int idx = blockIdx.x * blockDim.x + threadIdx.x;   // H_*NS_
  if (idx >= H_ * NS_) return;
  int hh = idx >> 6;
  int n  = idx & (NS_ - 1);
  float dt = __expf(log_dt[li * H_ + hh]);
  float A  = -__expf(A_log[((size_t)li * H_ + hh) * NS_ + n]);
  float da = dt * A;
  float Bbar = (__expf(da) - 1.0f) / A;
  dtA[idx] = da;
  CB[idx]  = C[((size_t)li * H_ + hh) * NS_ + n] * Bbar;
}

// ------------------- Vandermonde kernel K[h,l] (bf16) -----------------------
__global__ void s4d_kernelK(const float* __restrict__ dtA,
                            const float* __restrict__ CB,
                            __bf16* __restrict__ Kk) {
  __shared__ __attribute__((aligned(16))) float scoef[2 * NS_]; // [0..63]=dtA row, [64..127]=CB row
  int hh = blockIdx.x;
#ifdef HAVE_TDM
  if (threadIdx.x < 32) {
    // Tensor DMA: one descriptor moves the 2x64 f32 coefficient tile to LDS.
    unsigned long long ga = (unsigned long long)(uintptr_t)(dtA + (size_t)hh * NS_);
    unsigned int ldsoff = (unsigned int)(uintptr_t)&scoef[0];
    u32x4_t g0 = { 1u,                                   // count=1 (valid D#)
                   ldsoff,                               // lds_addr
                   (unsigned int)ga,                     // global_addr[31:0]
                   (unsigned int)((ga >> 32) & 0x1FFFFFFu) | (2u << 30) }; // addr[56:32] | type=2
    i32x8_t g1 = { (int)(2u << 16),                      // data_size=4B; mask=0; no flags
                   (int)(64u << 16),                     // tensor_dim0[15:0]=64 (bits63:48)
                   (int)(2u << 16),                      // tensor_dim1[15:0]=2  (bits95:80)
                   (int)(64u << 16),                     // tile_dim0=64 (bits127:112)
                   (int)2,                               // tile_dim1=2 (bits143:128)
                   (int)(H_ * NS_),                      // tensor_dim0_stride[31:0] (elements)
                   0, 0 };
    i32x4_t gz4 = { 0, 0, 0, 0 };
    i32x8_t gz8 = { 0, 0, 0, 0, 0, 0, 0, 0 };
    // 6-arg form (amdgpu-toolchain / clang-23 variant)
    __builtin_amdgcn_tensor_load_to_lds(g0, g1, gz4, gz4, gz8, 0);
    __builtin_amdgcn_s_wait_tensorcnt(0);
  }
#else
  if (threadIdx.x < NS_) {
    scoef[threadIdx.x]       = dtA[hh * NS_ + threadIdx.x];
    scoef[NS_ + threadIdx.x] = CB[hh * NS_ + threadIdx.x];
  }
#endif
  __syncthreads();
  int l = blockIdx.y * blockDim.x + threadIdx.x;     // grid.y*256 == L_
  float fl = (float)l;
  float s = 0.f;
#pragma unroll 8
  for (int n = 0; n < NS_; ++n) s += scoef[NS_ + n] * __expf(scoef[n] * fl);
  Kk[(size_t)hh * L_ + l] = (__bf16)s;
}

// ---------------- tiled transpose [BL][H] -> [H][BL] (bf16) -----------------
__global__ void transpose_kernel(const __bf16* __restrict__ in,
                                 __bf16* __restrict__ out) {
  __shared__ __bf16 tile[32][34];
  int r0 = blockIdx.y * 32;                          // BL dim
  int c0 = blockIdx.x * 32;                          // H dim
  int tx = threadIdx.x;                              // 0..31
  for (int i = threadIdx.y; i < 32; i += 8)
    tile[i][tx] = in[(size_t)(r0 + i) * H_ + (c0 + tx)];
  __syncthreads();
  for (int i = threadIdx.y; i < 32; i += 8)
    out[(size_t)(c0 + i) * BL_ + (r0 + tx)] = tile[tx][i];
}

// ---------------- S4D causal conv as per-channel Toeplitz WMMA --------------
// For channel h:  Y(32 x 1024) = X(32 x 1024) @ T_h,  T_h[j,l] = K[h, l-j].
// One wave computes a 32(batch) x 32(l) output block (4 accumulators):
//  - the two A-fragments are shared by both l-tiles,
//  - each Toeplitz B-fragment feeds two WMMAs,
//  - B-fragments come from a MIRRORED packed-pair array so the 8 dwords of a
//    fragment are at ascending LDS addresses (ds_load_2addr lands results
//    directly in fragment register order, no shuffle movs):
//      P[QTOP - i] = {lo:Kpad[i], hi:Kpad[i-1]}
//      fragment dword v = P[(QTOP - base) + 2v]
#define KPAD 64
#define QTOP (KPAD + L_ - 1)
__global__ void s4d_conv_kernel(const __bf16* __restrict__ hT,
                                const __bf16* __restrict__ Kk,
                                __bf16* __restrict__ out) {
  __shared__ __attribute__((aligned(16))) __bf16 ldsK[KPAD + L_];
  __shared__ unsigned int ldsQ[KPAD + L_];
  const int h = blockIdx.x;

  // stage Kpad = [64 zeros | K[h,0..1023]] into LDS
  for (int i = threadIdx.x; i < KPAD; i += blockDim.x) ldsK[i] = (__bf16)0.f;
#ifdef HAVE_ASYNC_LDS
  if (threadIdx.x < (L_ / 8)) {                      // 128 lanes x 16B = 2KB
    const __bf16* gsrc = Kk + (size_t)h * L_ + threadIdx.x * 8;
    __builtin_amdgcn_global_load_async_to_lds_b128(
        (gv4i*)(unsigned long long)(uintptr_t)gsrc,
        (sv4i*)(unsigned int)(uintptr_t)&ldsK[KPAD + threadIdx.x * 8],
        0, 0);
  }
  __builtin_amdgcn_s_wait_asynccnt(0);
#else
  for (int i = threadIdx.x; i < L_; i += blockDim.x)
    ldsK[KPAD + i] = Kk[(size_t)h * L_ + i];
#endif
  __syncthreads();

  // build mirrored packed descending pairs
  {
    const unsigned short* ks = (const unsigned short*)ldsK;
    for (int i = threadIdx.x; i < KPAD + L_; i += blockDim.x) {
      unsigned int lo = ks[i];
      unsigned int hi = (i > 0) ? ks[i - 1] : 0u;
      ldsQ[QTOP - i] = lo | (hi << 16);
    }
  }
  __syncthreads();

  const int wave = threadIdx.x >> 5;                 // 8 waves / block
  const int lane = threadIdx.x & 31;
  const int half = lane >> 4;
  const int lr   = lane & 15;

  int pr = blockIdx.y * (blockDim.x >> 5) + wave;    // l-pair index [0,32)
  int l0 = pr * 32;                                  // covers l-tiles l0, l0+16

  const __bf16* Xr0 = hT + (size_t)h * BL_ + (size_t)lr * L_;        // batch 0..15
  const __bf16* Xr1 = Xr0 + (size_t)16 * L_;                         // batch 16..31

  v8f acc00 = {}, acc01 = {}, acc10 = {}, acc11 = {};
  const int jmax = l0 + 31;                          // causal K-dim limit of the pair
  for (int j0 = 0; j0 <= jmax; j0 += 32) {
    Frag a0, a1;
    BFrag bA, bB;
    int ka = j0 + half * 16;                         // A layout: K 0..15 / 16..31
    a0.q[0] = *(const u32x4*)(Xr0 + ka);
    a0.q[1] = *(const u32x4*)(Xr0 + ka + 8);
    a1.q[0] = *(const u32x4*)(Xr1 + ka);
    a1.q[1] = *(const u32x4*)(Xr1 + ka + 8);
    int baseA = KPAD + (l0 + lr)      - j0 - half * 16;
    int qA = QTOP - baseA;                           // ascending reads
    int qB = qA - 16;                                // baseB = baseA + 16
#pragma unroll
    for (int v = 0; v < 8; ++v) {
      bA.u[v] = ldsQ[qA + 2 * v];
      bB.u[v] = ldsQ[qB + 2 * v];
    }
    acc00 = wmma_bf16(a0.v, bA.v, acc00);
    acc10 = wmma_bf16(a1.v, bA.v, acc10);
    acc01 = wmma_bf16(a0.v, bB.v, acc01);
    acc11 = wmma_bf16(a1.v, bB.v, acc11);
  }
#pragma unroll
  for (int r = 0; r < 8; ++r) {
    int row = r + half * 8;
    int lA  = l0 + lr;
    int lB  = l0 + 16 + lr;
    out[((size_t)row * L_ + lA) * H_ + h]        = (__bf16)acc00[r];
    out[((size_t)row * L_ + lB) * H_ + h]        = (__bf16)acc01[r];
    out[((size_t)(row + 16) * L_ + lA) * H_ + h] = (__bf16)acc10[r];
    out[((size_t)(row + 16) * L_ + lB) * H_ + h] = (__bf16)acc11[r];
  }
}

// ---------------- W_ff -> bf16 transposed (N x K per layer) -----------------
__global__ void wt_convert(const float* __restrict__ W_ff,
                           __bf16* __restrict__ WT) {
  int idx = blockIdx.x * blockDim.x + threadIdx.x;   // NL_*H_*H_
  if (idx >= NL_ * H_ * H_) return;
  int li  = idx / (H_ * H_);
  int rem = idx - li * (H_ * H_);
  int n = rem / H_;
  int k = rem - n * H_;
  WT[idx] = (__bf16)W_ff[((size_t)li * H_ + k) * H_ + n];
}

// ---------------- FF GEMM: Y = relu(X @ W) via WMMA -------------------------
// X: [BL][H] bf16 ; WT: [N=H][K=H] bf16 (transposed) ; Y: [BL][H] bf16
// One wave computes a 32(M) x 64(N) tile: 8 WMMAs per 12 b128 loads.
__global__ void ff_kernel(const __bf16* __restrict__ X,
                          const __bf16* __restrict__ WT,
                          __bf16* __restrict__ Y) {
  const int wave = threadIdx.x >> 5;                 // 4 waves / block
  const int lane = threadIdx.x & 31;
  const int half = lane >> 4;
  const int lr   = lane & 15;

  int m0 = blockIdx.x * 32;
  int n0 = (blockIdx.y * (blockDim.x >> 5) + wave) * 64;   // 8 n-groups of 64

  const __bf16* Xr0 = X + (size_t)(m0 + lr) * H_;
  const __bf16* Xr1 = X + (size_t)(m0 + 16 + lr) * H_;
  const __bf16* Wr0 = WT + (size_t)(n0 + 0  + lr) * H_;
  const __bf16* Wr1 = WT + (size_t)(n0 + 16 + lr) * H_;
  const __bf16* Wr2 = WT + (size_t)(n0 + 32 + lr) * H_;
  const __bf16* Wr3 = WT + (size_t)(n0 + 48 + lr) * H_;

  v8f acc00 = {}, acc01 = {}, acc02 = {}, acc03 = {};
  v8f acc10 = {}, acc11 = {}, acc12 = {}, acc13 = {};
  for (int k0 = 0; k0 < H_; k0 += 32) {
    Frag a0, a1, b0, b1, b2, b3;
    int ka = k0 + half * 16;
    a0.q[0] = *(const u32x4*)(Xr0 + ka);
    a0.q[1] = *(const u32x4*)(Xr0 + ka + 8);
    a1.q[0] = *(const u32x4*)(Xr1 + ka);
    a1.q[1] = *(const u32x4*)(Xr1 + ka + 8);
    __builtin_prefetch(Xr0 + k0 + 128, 0, 1);        // stream-ahead on X
    int kb = k0 + half * 16;                         // B layout: K 0..15 / 16..31
    b0.q[0] = *(const u32x4*)(Wr0 + kb);
    b0.q[1] = *(const u32x4*)(Wr0 + kb + 8);
    b1.q[0] = *(const u32x4*)(Wr1 + kb);
    b1.q[1] = *(const u32x4*)(Wr1 + kb + 8);
    b2.q[0] = *(const u32x4*)(Wr2 + kb);
    b2.q[1] = *(const u32x4*)(Wr2 + kb + 8);
    b3.q[0] = *(const u32x4*)(Wr3 + kb);
    b3.q[1] = *(const u32x4*)(Wr3 + kb + 8);
    acc00 = wmma_bf16(a0.v, b0.v, acc00);
    acc01 = wmma_bf16(a0.v, b1.v, acc01);
    acc02 = wmma_bf16(a0.v, b2.v, acc02);
    acc03 = wmma_bf16(a0.v, b3.v, acc03);
    acc10 = wmma_bf16(a1.v, b0.v, acc10);
    acc11 = wmma_bf16(a1.v, b1.v, acc11);
    acc12 = wmma_bf16(a1.v, b2.v, acc12);
    acc13 = wmma_bf16(a1.v, b3.v, acc13);
  }
#pragma unroll
  for (int r = 0; r < 8; ++r) {
    size_t ro0 = (size_t)(m0 + r + half * 8) * H_;
    size_t ro1 = (size_t)(m0 + 16 + r + half * 8) * H_;
    Y[ro0 + n0 + 0  + lr] = (__bf16)fmaxf(acc00[r], 0.0f);
    Y[ro0 + n0 + 16 + lr] = (__bf16)fmaxf(acc01[r], 0.0f);
    Y[ro0 + n0 + 32 + lr] = (__bf16)fmaxf(acc02[r], 0.0f);
    Y[ro0 + n0 + 48 + lr] = (__bf16)fmaxf(acc03[r], 0.0f);
    Y[ro1 + n0 + 0  + lr] = (__bf16)fmaxf(acc10[r], 0.0f);
    Y[ro1 + n0 + 16 + lr] = (__bf16)fmaxf(acc11[r], 0.0f);
    Y[ro1 + n0 + 32 + lr] = (__bf16)fmaxf(acc12[r], 0.0f);
    Y[ro1 + n0 + 48 + lr] = (__bf16)fmaxf(acc13[r], 0.0f);
  }
}

// ---------------- decode at last position -----------------------------------
__global__ void decode_kernel(const __bf16* __restrict__ hfin,
                              const float* __restrict__ W_dec,
                              float* __restrict__ out) {
  int t = threadIdx.x;
  if (t >= B_ * DOUT_) return;
  int b = t / DOUT_;
  int o = t - b * DOUT_;
  const __bf16* hp = hfin + ((size_t)b * L_ + (L_ - 1)) * H_;
  float s = 0.f;
  for (int k = 0; k < H_; ++k) s += (float)hp[k] * W_dec[(size_t)k * DOUT_ + o];
  out[t] = s;
}

// ---------------------------------------------------------------------------
extern "C" void kernel_launch(void* const* d_in, const int* in_sizes, int n_in,
                              void* d_out, int out_size, void* d_ws, size_t ws_size,
                              hipStream_t stream) {
  (void)in_sizes; (void)n_in; (void)out_size; (void)ws_size;
  const float* x      = (const float*)d_in[0];
  const float* W_enc  = (const float*)d_in[1];
  const float* log_dt = (const float*)d_in[2];
  const float* A_log  = (const float*)d_in[3];
  const float* C      = (const float*)d_in[4];
  const float* W_ff   = (const float*)d_in[5];
  const float* W_dec  = (const float*)d_in[6];

  char* ws = (char*)d_ws;
  const size_t ACT = (size_t)BL_ * H_ * sizeof(__bf16);   // 33.5 MB
  __bf16* bufA = (__bf16*)(ws);                            // activations (BLH)
  __bf16* bufB = (__bf16*)(ws + ACT);                      // conv out (BLH)
  __bf16* bufT = (__bf16*)(ws + 2 * ACT);                  // transposed (HBL)
  char*   p    = ws + 3 * ACT;
  __bf16* WT   = (__bf16*)p;  p += (size_t)NL_ * H_ * H_ * sizeof(__bf16);
  __bf16* Kk   = (__bf16*)p;  p += (size_t)H_ * L_ * sizeof(__bf16);
  float*  dtA  = (float*)p;   p += (size_t)H_ * NS_ * sizeof(float);
  float*  CB   = (float*)p;   // contiguous after dtA (TDM reads both as one tile)

  wt_convert<<<dim3((NL_ * H_ * H_) / 256), dim3(256), 0, stream>>>(W_ff, WT);
  encode_kernel<<<dim3((BL_ * H_) / 256), dim3(256), 0, stream>>>(x, W_enc, bufA);

  for (int li = 0; li < NL_; ++li) {
    coef_kernel<<<dim3((H_ * NS_) / 256), dim3(256), 0, stream>>>(log_dt, A_log, C, dtA, CB, li);
    s4d_kernelK<<<dim3(H_, L_ / 256), dim3(256), 0, stream>>>(dtA, CB, Kk);
    transpose_kernel<<<dim3(H_ / 32, BL_ / 32), dim3(32, 8), 0, stream>>>(bufA, bufT);
    s4d_conv_kernel<<<dim3(H_, 4), dim3(256), 0, stream>>>(bufT, Kk, bufB);
    ff_kernel<<<dim3(BL_ / 32, 2), dim3(128), 0, stream>>>(
        bufB, WT + (size_t)li * H_ * H_, bufA);
  }

  decode_kernel<<<dim3(1), dim3(512), 0, stream>>>(bufA, W_dec, (float*)d_out);
}
